// Attention_10505490006301
// MI455X (gfx1250) — matrix operand order repaired and checked by reading, not compiled, not run on hardware
//
#include <hip/hip_runtime.h>
#include <hip/hip_bf16.h>
#include <math.h>

#define B_   32
#define S_   2048
#define ENC_ 1024
#define DEC_ 1024

typedef __bf16 v16bf __attribute__((ext_vector_type(16)));
typedef float  v8f   __attribute__((ext_vector_type(8)));
typedef unsigned int v4u __attribute__((ext_vector_type(4)));
typedef int v8i __attribute__((ext_vector_type(8)));
typedef int v4i __attribute__((ext_vector_type(4)));

union Frag16 { v16bf v; uint4 q[2]; };

// ---------------- kernel 0: w_enc (ENC x DEC, f32) -> wT[n][k] bf16 (DEC x ENC) ----------------
__global__ void wprep_kernel(const float* __restrict__ a_w, __hip_bfloat16* __restrict__ wT) {
  const int e = blockIdx.x;                       // k index 0..ENC-1
  for (int j = 0; j < 4; ++j) {
    const int n = threadIdx.x + j * 256;          // DEC column
    wT[(size_t)n * ENC_ + e] = __float2bfloat16(a_w[(size_t)(DEC_ + e) * DEC_ + n]);
  }
}

// ---------------- kernel 1: hAll[b][d] = hidden[b] @ w_dec + a_b ----------------
__global__ void hproj_kernel(const float* __restrict__ hidden, const float* __restrict__ a_w,
                             const float* __restrict__ a_b, float* __restrict__ hAll) {
  const int b = blockIdx.x;
  __shared__ float hrow[DEC_];
  for (int j = 0; j < 4; ++j)
    hrow[threadIdx.x + j * 256] = hidden[(size_t)b * DEC_ + threadIdx.x + j * 256];
  __syncthreads();
  for (int j = 0; j < 4; ++j) {
    const int d = threadIdx.x + j * 256;
    float acc = a_b[d];
    for (int k = 0; k < DEC_; ++k)
      acc = fmaf(hrow[k], a_w[(size_t)k * DEC_ + d], acc);
    hAll[(size_t)b * DEC_ + d] = acc;
  }
}

// ---------------- kernel 2: fused e_proj GEMM (bf16 WMMA) + tanh + dot(v_w) -> scores ----------
#define A_STR  1032   // bf16 elems per A row in LDS (2064B, 16B-aligned, bank-staggered)
#define W_STR  72     // bf16 elems per W row in LDS (144B = 32 data dwords + 4 pad dwords)
#define W_ELEM (128 * W_STR)       // one W ping/pong buffer, in bf16 elements
#define W_BYTES (W_ELEM * 2)

__device__ __forceinline__ void tdm_issue_w(const __hip_bfloat16* gsrc, unsigned lds_addr) {
  // TDM: load 128(n) x 64(k) bf16 tile into LDS with 4-dword row padding (D# per ISA 08 §8)
  const unsigned long long ga = (unsigned long long)(uintptr_t)gsrc;
  v4u g0; v8i g1; v4i gz4 = {0, 0, 0, 0};
  v8i gz8 = {0, 0, 0, 0, 0, 0, 0, 0};
  g0[0] = 1u;                                    // count=1, gather off
  g0[1] = lds_addr;                              // lds_addr
  g0[2] = (unsigned)ga;                          // global_addr[31:0]
  g0[3] = (unsigned)((ga >> 32) & 0x01FFFFFFull) | (2u << 30); // addr[56:32] | type=2
  g1[0] = (1 << 16) | (1 << 20) | (4 << 22) | (3 << 25);
           // data_size=2B | pad_enable | pad_interval=32dw | pad_amount=4dw
  g1[1] = (1024 << 16);                          // tensor_dim0 = 1024
  g1[2] = (128 << 16);                           // tensor_dim1 = 128
  g1[3] = (64 << 16);                            // tile_dim0 = 64
  g1[4] = 128;                                   // tile_dim1 = 128, tile_dim2 = 0
  g1[5] = 1024;                                  // tensor_dim0_stride = 1024 elements
  g1[6] = 0;
  g1[7] = 0;
  __builtin_amdgcn_tensor_load_to_lds(g0, g1, gz4, gz4, gz8, 0);
}

__global__ __launch_bounds__(256) void scores_kernel(
    const float* __restrict__ enc, const __hip_bfloat16* __restrict__ wT,
    const float* __restrict__ hAll, const float* __restrict__ v_w,
    float* __restrict__ scores) {
  const int b   = blockIdx.x >> 7;          // S_/16 = 128 tiles per batch
  const int s0  = (blockIdx.x & 127) << 4;
  const int t   = threadIdx.x;
  const int lane  = t & 31;
  const int wv    = t >> 5;                 // wave 0..7, owns 16 N-columns
  const int lrow  = lane & 15;
  const int lhalf = lane >> 4;

  __shared__ __align__(16) __hip_bfloat16 Alds[16 * A_STR];
  __shared__ __align__(16) __hip_bfloat16 Wlds[2 * W_ELEM];   // ping/pong W buffers
  __shared__ float sred[16];

  if (t < 16) sred[t] = 0.f;

  // stage A tile: 16 S-rows x 1024 K, f32 -> bf16
  for (int i = 0; i < 64; ++i) {
    const int idx = t + i * 256;
    const int m = idx >> 10, k = idx & 1023;
    Alds[m * A_STR + k] = __float2bfloat16(enc[((size_t)b * S_ + s0 + m) * (size_t)ENC_ + k]);
  }

  const unsigned ldsW0 = (unsigned)(uintptr_t)&Wlds[0];
  float part[8] = {0.f, 0.f, 0.f, 0.f, 0.f, 0.f, 0.f, 0.f};
  int pg = 0;                                // ping/pong selector (uniform)

  // prologue: start DMA of global chunk 0 into buffer 0
  if (wv == 0) tdm_issue_w(wT, ldsW0);

  for (int np = 0; np < 8; ++np) {           // 8 passes of 128 N-columns
    v8f acc = {};
    for (int ci = 0; ci < 16; ++ci) {        // K chunks of 64
      const int g = np * 16 + ci;            // global chunk index 0..127
      if (wv == 0) __builtin_amdgcn_s_wait_tensorcnt(0);   // chunk g landed in buf[pg]
      __syncthreads();                       // publish buf[pg]; all reads of buf[pg^1] done
      if (wv == 0 && g + 1 < 128) {
        const int gn = g + 1;
        tdm_issue_w(wT + ((size_t)(gn >> 4) * 128) * ENC_ + (gn & 15) * 64,
                    ldsW0 + (unsigned)((pg ^ 1) * W_BYTES));  // overlap DMA with compute
      }
      const __hip_bfloat16* wb = &Wlds[pg * W_ELEM];
      const int kc = ci * 64;
      #pragma unroll
      for (int ks = 0; ks < 2; ++ks) {       // two K=32 WMMA steps per chunk
        const int k0 = kc + ks * 32;
        Frag16 af, bfr;
        // A 16x32 bf16 frag: elems 0..7 -> K = 8*lhalf..+7 ; elems 8..15 -> K = 16+8*lhalf..+7
        const __hip_bfloat16* ap = &Alds[lrow * A_STR + k0 + 8 * lhalf];
        af.q[0] = *(const uint4*)ap;
        af.q[1] = *(const uint4*)(ap + 16);
        // B 32x16 bf16 frag: col n = lrow, elems 0..7 -> K = 8*lhalf..+7 ; elems 8..15 -> K = 16+8*lhalf..+7
        const __hip_bfloat16* bp = &wb[(wv * 16 + lrow) * W_STR + (ks * 32 + 8 * lhalf)];
        bfr.q[0] = *(const uint4*)bp;
        bfr.q[1] = *(const uint4*)(bp + 16);
        acc = __builtin_amdgcn_wmma_f32_16x16x32_bf16(
            false, af.v, false, bfr.v, (short)0, acc, false, false);
      }
      pg ^= 1;
    }
    // epilogue: energy = tanh(e + h_proj + a_b); partial score += energy * v_w[n]
    const int n   = np * 128 + wv * 16 + lrow;       // C layout: N = lane&15
    const float hb = hAll[(size_t)b * DEC_ + n];
    const float vw = v_w[n];
    #pragma unroll
    for (int r = 0; r < 8; ++r)                      // C layout: M = r + 8*lhalf
      part[r] += __builtin_amdgcn_tanhf(acc[r] + hb) * vw;   // hardware V_TANH_F32
  }

  // reduce the 16 N-lanes sharing lhalf, then combine waves through LDS
  #pragma unroll
  for (int r = 0; r < 8; ++r) {
    float v = part[r];
    for (int m = 8; m >= 1; m >>= 1) v += __shfl_xor(v, m, 16);
    if (lrow == 0) atomicAdd(&sred[r + 8 * lhalf], v);
  }
  __syncthreads();
  if (t < 16) scores[(size_t)b * S_ + s0 + t] = sred[t];
}

// ---------------- kernel 3: masked softmax over S, and zero the output for atomics ------------
__global__ void softmax_kernel(const float* __restrict__ scores, const int* __restrict__ masks,
                               float* __restrict__ attn, float* __restrict__ out) {
  const int b = blockIdx.x, t = threadIdx.x;
  __shared__ float red[8];
  float v[8];
  float lmax = -3.0e38f;
  for (int i = 0; i < 8; ++i) {
    const int s = t + i * 256;
    float x = scores[(size_t)b * S_ + s];
    if (masks[(size_t)b * S_ + s] == 0) x = -1.0e10f;
    v[i] = x;
    lmax = fmaxf(lmax, x);
  }
  for (int m = 16; m >= 1; m >>= 1) lmax = fmaxf(lmax, __shfl_xor(lmax, m, 32));
  if ((t & 31) == 0) red[t >> 5] = lmax;
  __syncthreads();
  float bmax = red[0];
  for (int j = 1; j < 8; ++j) bmax = fmaxf(bmax, red[j]);
  float lsum = 0.f;
  for (int i = 0; i < 8; ++i) { v[i] = __expf(v[i] - bmax); lsum += v[i]; }
  for (int m = 16; m >= 1; m >>= 1) lsum += __shfl_xor(lsum, m, 32);
  __syncthreads();
  if ((t & 31) == 0) red[t >> 5] = lsum;
  __syncthreads();
  float bsum = 0.f;
  for (int j = 0; j < 8; ++j) bsum += red[j];
  const float inv = 1.0f / bsum;
  for (int i = 0; i < 8; ++i) attn[(size_t)b * S_ + t + i * 256] = v[i] * inv;
  for (int j = 0; j < 4; ++j) out[(size_t)b * ENC_ + t + j * 256] = 0.f;   // for atomic pass
}

// ---------------- kernel 4: weighted[b][e] = sum_s attn[b][s] * enc[b][s][e] ------------------
__global__ void weighted_kernel(const float* __restrict__ attn, const float* __restrict__ enc,
                                float* __restrict__ out) {
  const int b = blockIdx.x, sc = blockIdx.y, t = threadIdx.x;
  __shared__ float aL[256];
  aL[t] = attn[(size_t)b * S_ + sc * 256 + t];
  __syncthreads();
  float acc[4] = {0.f, 0.f, 0.f, 0.f};
  const float* ebase = enc + ((size_t)b * S_ + (size_t)sc * 256) * ENC_;
  for (int s = 0; s < 256; ++s) {
    const float a = aL[s];
    const float* row = ebase + (size_t)s * ENC_;
    if ((s & 15) == 0) __builtin_prefetch(row + 4 * ENC_, 0, 0);
    #pragma unroll
    for (int j = 0; j < 4; ++j) acc[j] = fmaf(a, row[t + j * 256], acc[j]);
  }
  #pragma unroll
  for (int j = 0; j < 4; ++j) atomicAdd(&out[(size_t)b * ENC_ + t + j * 256], acc[j]);
}

// ---------------- launch ----------------
extern "C" void kernel_launch(void* const* d_in, const int* in_sizes, int n_in,
                              void* d_out, int out_size, void* d_ws, size_t ws_size,
                              hipStream_t stream) {
  (void)in_sizes; (void)n_in; (void)out_size; (void)ws_size;
  const float* hidden = (const float*)d_in[0];
  const float* enc    = (const float*)d_in[1];
  const int*   masks  = (const int*)d_in[2];
  const float* a_w    = (const float*)d_in[3];
  const float* a_b    = (const float*)d_in[4];
  const float* v_w    = (const float*)d_in[5];
  float* out = (float*)d_out;

  // workspace carve-out: wT (2MB bf16) | hAll (128KB) | scores (256KB) | attn (256KB)
  __hip_bfloat16* wT = (__hip_bfloat16*)d_ws;
  float* hAll   = (float*)((char*)d_ws + (2u << 20));
  float* scores = hAll + (size_t)B_ * DEC_;
  float* attn   = scores + (size_t)B_ * S_;

  wprep_kernel  <<<ENC_, 256, 0, stream>>>(a_w, wT);
  hproj_kernel  <<<B_, 256, 0, stream>>>(hidden, a_w, a_b, hAll);
  scores_kernel <<<B_ * (S_ / 16), 256, 0, stream>>>(enc, wT, hAll, v_w, scores);
  softmax_kernel<<<B_, 256, 0, stream>>>(scores, masks, attn, out);
  weighted_kernel<<<dim3(B_, S_ / 256), 256, 0, stream>>>(attn, enc, out);
}
